// Hydra_13846974562838
// MI455X (gfx1250) — matrix operand order repaired and checked by reading, not compiled, not run on hardware
//
#include <hip/hip_runtime.h>
#include <cstdint>

// ---------------------------------------------------------------------------
// Hydra (Mamba2-SSD + attention fusion) forward for MI455X (gfx1250, wave32).
// Dense projections: v_wmma_f32_16x16x32_f16 fed by TENSOR_LOAD_TO_LDS (TDM)
// with double-buffered LDS (DMA overlaps WMMA; s_wait_tensorcnt sync).
// Shapes: B=2 L=256 Dm=512 Din=1024 H=32 P=32 N=64 Dproj=2208 E=64 BH=64
// ---------------------------------------------------------------------------

typedef _Float16 half16 __attribute__((ext_vector_type(16)));
typedef _Float16 half8  __attribute__((ext_vector_type(8)));
typedef _Float16 half4  __attribute__((ext_vector_type(4)));
typedef float    floatx8 __attribute__((ext_vector_type(8)));
typedef unsigned int v4u __attribute__((ext_vector_type(4)));
typedef int          v8i __attribute__((ext_vector_type(8)));
typedef int          v4i __attribute__((ext_vector_type(4)));

union F16x16 { half16 v; half8 h[2]; };

#define GB 2
#define GL 256
#define GH 32
#define GP 32
#define GN 64
#define GDM 512
#define GDIN 1024
#define GDPROJ 2208

#if __has_builtin(__builtin_amdgcn_tensor_load_to_lds) && \
    __has_builtin(__builtin_amdgcn_s_wait_tensorcnt)
#define USE_TDM 1
#else
#define USE_TDM 0
#endif

#if USE_TDM
// ---------------------------------------------------------------------------
// Issue one TDM descriptor: load a (rows x 32) f16 tile (row stride `stride`
// elements) from global into LDS at byte offset lds_off (packed row-major).
// D# packing per CDNA5 ISA ch.8: group0 {count=1, lds_addr, global_addr,
// type=2}; group1 {data_size=1(2B), tensor_dim0=stride, tensor_dim1=rows,
// tile_dim0=32, tile_dim1=rows, tensor_dim0_stride=stride}.
// ---------------------------------------------------------------------------
__device__ static inline void tdm_load_tile_f16(uint32_t lds_off,
                                                const _Float16* gptr,
                                                uint32_t rows, uint32_t stride)
{
  uint64_t ga = (uint64_t)(uintptr_t)gptr;
  v4u g0 = {0u, 0u, 0u, 0u};
  g0[0] = 1u;                                   // count=1, gather off
  g0[1] = lds_off;                              // lds_addr (bytes)
  g0[2] = (uint32_t)ga;                         // global_addr[31:0]
  g0[3] = (uint32_t)((ga >> 32) & 0x01FFFFFFu)  // global_addr[56:32]
          | 0x80000000u;                        // type=2 ("image")
  v8i g1 = {0, 0, 0, 0, 0, 0, 0, 0};
  g1[0] = 0x00010000;                           // data_size = 2 bytes
  g1[1] = (int)((stride & 0xffffu) << 16);      // tensor_dim0 lo -> [63:48]
  g1[2] = (int)((stride >> 16) | ((rows & 0xffffu) << 16)); // dim0 hi | dim1 lo
  g1[3] = (int)((rows >> 16) | (32u << 16));    // dim1 hi | tile_dim0 = 32
  g1[4] = (int)rows;                            // tile_dim1 = rows
  g1[5] = (int)stride;                          // tensor_dim0_stride lo
  v4i z4 = {0, 0, 0, 0};
#if defined(__clang_major__) && __clang_major__ >= 23
  v8i z8 = {0, 0, 0, 0, 0, 0, 0, 0};
  __builtin_amdgcn_tensor_load_to_lds(g0, g1, z4, z4, z8, 0);
#else
  __builtin_amdgcn_tensor_load_to_lds(g0, g1, z4, z4, 0);
#endif
}
#endif  // USE_TDM

// ---------------------------------------------------------------------------
// Generic f16 WMMA GEMM: C[M,N] = A[M,K] @ B[N,K]^T (+ bias[N]), f32 out.
// Block tile 64x32, 8 waves (4x2 wave grid of 16x16 WMMA tiles), K step 32.
// TDM path: wave 0 DMAs the next K-tile into the spare LDS buffer while all
// waves compute; s_wait_tensorcnt + one barrier per K-step.
// Requires M%64==0, N%32==0, K%32==0 (true for every call site here).
// ---------------------------------------------------------------------------
__global__ __launch_bounds__(256) void wmma_gemm_f16(
    const _Float16* __restrict__ A, const _Float16* __restrict__ B,
    const float* __restrict__ bias, float* __restrict__ C,
    int M, int N, int K, int lda, int ldb, int ldc,
    long batchA, long batchB, long batchC)
{
  __shared__ __align__(16) _Float16 As[2][64][32];
  __shared__ __align__(16) _Float16 Bs[2][32][32];

  const int bz = blockIdx.z;
  A += (long)bz * batchA;
  B += (long)bz * batchB;
  C += (long)bz * batchC;

  const int m0   = blockIdx.y * 64;
  const int n0   = blockIdx.x * 32;
  const int tid  = threadIdx.x;
  const int lane = tid & 31;
  const int wave = tid >> 5;
  const int wm   = wave >> 1;   // 0..3
  const int wn   = wave & 1;    // 0..1

  floatx8 acc = {0.f, 0.f, 0.f, 0.f, 0.f, 0.f, 0.f, 0.f};

#if USE_TDM
  const uint32_t ldsA0 = (uint32_t)(uintptr_t)&As[0][0][0];
  const uint32_t ldsA1 = (uint32_t)(uintptr_t)&As[1][0][0];
  const uint32_t ldsB0 = (uint32_t)(uintptr_t)&Bs[0][0][0];
  const uint32_t ldsB1 = (uint32_t)(uintptr_t)&Bs[1][0][0];
  if (wave == 0) {  // preload K-tile 0 into buffer 0 (TDM ignores EXEC)
    tdm_load_tile_f16(ldsA0, &A[(long)m0 * lda], 64u, (uint32_t)lda);
    tdm_load_tile_f16(ldsB0, &B[(long)n0 * ldb], 32u, (uint32_t)ldb);
    __builtin_amdgcn_s_wait_tensorcnt(0);
  }
  __syncthreads();
#else
  const int arow = tid >> 2;        // 0..63
  const int acol = (tid & 3) * 8;   // 0,8,16,24
  const int brow = tid >> 3;        // 0..31
  const int bcol = (tid & 7) * 4;   // 0..28
  *(half8*)&As[0][arow][acol] =
      *(const half8*)&A[(long)(m0 + arow) * lda + acol];
  *(half4*)&Bs[0][brow][bcol] =
      *(const half4*)&B[(long)(n0 + brow) * ldb + bcol];
  __syncthreads();
#endif

  int buf = 0;
  for (int k0 = 0; k0 < K; k0 += 32) {
    const int nxt = buf ^ 1;
#if USE_TDM
    if (k0 + 32 < K && wave == 0) {  // DMA next K-tile while computing
      tdm_load_tile_f16(nxt ? ldsA1 : ldsA0,
                        &A[(long)m0 * lda + k0 + 32], 64u, (uint32_t)lda);
      tdm_load_tile_f16(nxt ? ldsB1 : ldsB0,
                        &B[(long)n0 * ldb + k0 + 32], 32u, (uint32_t)ldb);
    }
#else
    if (k0 + 32 < K) {
      __builtin_prefetch(&A[(long)(m0 + arow) * lda + k0 + 32 + acol], 0, 1);
      __builtin_prefetch(&B[(long)(n0 + brow) * ldb + k0 + 32 + bcol], 0, 1);
      *(half8*)&As[nxt][arow][acol] =
          *(const half8*)&A[(long)(m0 + arow) * lda + k0 + 32 + acol];
      *(half4*)&Bs[nxt][brow][bcol] =
          *(const half4*)&B[(long)(n0 + brow) * ldb + k0 + 32 + bcol];
    }
#endif

    // A fragment: lane (m = lane&15, khalf = lane>>4):
    //   elems 0..7  -> K = khalf*8 + 0..7 ; elems 8..15 -> K = 16+khalf*8+0..7
    const int am = wm * 16 + (lane & 15);
    const int kh = lane >> 4;
    F16x16 af, bf;
    af.h[0] = *(const half8*)&As[buf][am][kh * 8];
    af.h[1] = *(const half8*)&As[buf][am][16 + kh * 8];
    // B fragment: lane (n = lane&15, khalf): same K order as A
    const int bn = wn * 16 + (lane & 15);
    bf.h[0] = *(const half8*)&Bs[buf][bn][kh * 8];
    bf.h[1] = *(const half8*)&Bs[buf][bn][16 + kh * 8];

    acc = __builtin_amdgcn_wmma_f32_16x16x32_f16(
        false, af.v, false, bf.v, (short)0, acc, false, false);

#if USE_TDM
    if (wave == 0) __builtin_amdgcn_s_wait_tensorcnt(0);
#endif
    __syncthreads();
    buf = nxt;
  }

  const int cn    = n0 + wn * 16 + (lane & 15);
  const int mbase = m0 + wm * 16 + (lane >> 4) * 8;
  const float bv  = bias ? bias[cn] : 0.f;
#pragma unroll
  for (int r = 0; r < 8; ++r)
    C[(long)(mbase + r) * ldc + cn] = acc[r] + bv;
}

// ---------------------------------------------------------------------------
// f32 -> f16 conversion
// ---------------------------------------------------------------------------
__global__ void cvt_f32_f16(const float* __restrict__ s,
                            _Float16* __restrict__ d, long n)
{
  long i = (long)blockIdx.x * 256 + threadIdx.x;
  if (i < n) d[i] = (_Float16)s[i];
}

// ---------------------------------------------------------------------------
// prep: from zxbcdt compute Adt[b,t,h], xdt[b,t,h,p] (= x * dt),
// Bf16/Cf16 (f16 copies of B,C features for the Gram GEMM).
// zxbcdt row: z[0:1024] | x[1024:2048] | B[2048:2112] | C[2112:2176]
//             | dt_raw[2176:2208]
// ---------------------------------------------------------------------------
__global__ __launch_bounds__(256) void prep_kernel(
    const float* __restrict__ zx, const float* __restrict__ dt_bias,
    const float* __restrict__ A_log, float* __restrict__ Adt,
    float* __restrict__ xdt, _Float16* __restrict__ Bf16,
    _Float16* __restrict__ Cf16)
{
  const int t = blockIdx.x, b = blockIdx.y, tid = threadIdx.x;
  const float* row = zx + ((long)(b * GL) + t) * GDPROJ;
#pragma unroll
  for (int r = 0; r < 4; ++r) {
    int i = tid + 256 * r;
    int h = i >> 5;
    float raw = row[2176 + h] + dt_bias[h];
    float dtv = (raw > 20.f) ? raw : log1pf(__expf(raw));
    xdt[(((long)(b * GL) + t) * GH + h) * GP + (i & 31)] =
        row[1024 + i] * dtv;
    if ((i & 31) == 0)
      Adt[((b * GL) + t) * GH + h] = -__expf(A_log[h]) * dtv;
  }
  if (tid < 64)
    Bf16[((b * GL) + t) * GN + tid] = (_Float16)row[2048 + tid];
  else if (tid < 128)
    Cf16[((b * GL) + t) * GN + (tid - 64)] = (_Float16)row[2112 + (tid - 64)];
}

// ---------------------------------------------------------------------------
// Y_off (fused cross-chunk recurrence): per (b,h), thread z accumulates over
// c' = 1..255:
//   W[z,c'] = (c'<=z) ? exp(clip(padA_z*min(c',z-1),-50,50)) : 0
//   Yoff[z,l,:] += W * ( ee[c']*GT[b,c'-1,z+l]*xdt[c'-1,:]
//                        + GT[b,c',z+l]*xdt[c',:] )
// l=1 scaled by sdo = exp(Adt[b,z+1,h]).  GT[b][j][i] = Bf[b,j,:].Cf[b,i,:]
// ---------------------------------------------------------------------------
__global__ __launch_bounds__(256) void yoff_kernel(
    const float* __restrict__ GT, const float* __restrict__ xdt,
    const float* __restrict__ Adt, float* __restrict__ Yoff)
{
  const int b = blockIdx.x;
  const int h = blockIdx.y;
  const int z = threadIdx.x;  // 0..255 (z==255 produces no window)

  float acc0[GP], acc1[GP];
#pragma unroll
  for (int p = 0; p < GP; ++p) { acc0[p] = 0.f; acc1[p] = 0.f; }

  float padA = 0.f;
  if (z >= 1)
    padA = Adt[((b * GL) + z - 1) * GH + h] + Adt[((b * GL) + z) * GH + h];

  const float* xbase = xdt + (long)b * GL * GH * GP + (long)h * GP;
  const float* gbase = GT + (long)b * GL * GL;

  float xprev[GP];
#pragma unroll
  for (int p = 0; p < GP; ++p) xprev[p] = xbase[p];  // xdt[b,0,h,:]
  float g0prev = gbase[z];
  float g1prev = (z < GL - 1) ? gbase[z + 1] : 0.f;

  for (int cp = 1; cp < GL; ++cp) {
    float ee = __expf(Adt[((b * GL) + cp) * GH + h]);
    const float* grow = gbase + (long)cp * GL;
    float g0cur = grow[z];
    float g1cur = (z < GL - 1) ? grow[z + 1] : 0.f;

    float wgt = 0.f;
    if (cp <= z) {
      int mcz = (cp < z - 1) ? cp : (z - 1);
      float arg = padA * (float)mcz;
      arg = fminf(fmaxf(arg, -50.f), 50.f);
      wgt = __expf(arg);
    }
    float s00 = wgt * ee * g0prev;
    float s01 = wgt * g0cur;
    float s10 = wgt * ee * g1prev;
    float s11 = wgt * g1cur;

    const float* xcur = xbase + (long)cp * GH * GP;
#pragma unroll
    for (int p = 0; p < GP; ++p) {
      float x1 = xcur[p];
      acc0[p] += s00 * xprev[p] + s01 * x1;
      acc1[p] += s10 * xprev[p] + s11 * x1;
      xprev[p] = x1;
    }
    g0prev = g0cur;
    g1prev = g1cur;
  }

  if (z <= GL - 2) {
    float sdo1 = __expf(Adt[((b * GL) + z + 1) * GH + h]);
    float* o0 = Yoff + ((((long)(b * GL + z)) * 2 + 0) * GH + h) * GP;
    float* o1 = Yoff + ((((long)(b * GL + z)) * 2 + 1) * GH + h) * GP;
#pragma unroll
    for (int p = 0; p < GP; ++p) { o0[p] = acc0[p]; o1[p] = acc1[p] * sdo1; }
  }
}

// ---------------------------------------------------------------------------
// Restore (window -> sequence averaging) + build comb16 = [Yd | Yo] (f16).
// Yd uses the L_mat == [[1,0],[1,1]] identity-collapse:
//   Yd[t] = ([t<=L-2] G[t,t]*x[t] + [t>=1](G[t-1,t]*x[t-1] + G[t,t]*x[t]))/cnt
// ---------------------------------------------------------------------------
__global__ void combine_kernel(const float* __restrict__ GT,
                               const float* __restrict__ xdt,
                               const float* __restrict__ Yoff,
                               _Float16* __restrict__ comb)
{
  const int t = blockIdx.x, h = blockIdx.y, b = blockIdx.z;
  const int e = threadIdx.x;  // 0..63
  const float cnt = (t == 0 || t == GL - 1) ? 1.f : 2.f;
  float v = 0.f;
  if (e < 32) {
    int p = e;
    float xt  = xdt[(((long)(b * GL) + t) * GH + h) * GP + p];
    float gtt = GT[(((long)b * GL) + t) * GL + t];
    if (t <= GL - 2) v += gtt * xt;
    if (t >= 1) {
      float xm  = xdt[(((long)(b * GL) + t - 1) * GH + h) * GP + p];
      float g10 = GT[(((long)b * GL) + t - 1) * GL + t];
      v += g10 * xm + gtt * xt;
    }
  } else {
    int p = e - 32;
    if (t <= GL - 2)
      v += Yoff[((((long)(b * GL + t)) * 2 + 0) * GH + h) * GP + p];
    if (t >= 1)
      v += Yoff[((((long)(b * GL + t - 1)) * 2 + 1) * GH + h) * GP + p];
  }
  v /= cnt;
  comb[((((long)b * GH) + h) * GL + t) * 64 + e] = (_Float16)v;
}

// ---------------------------------------------------------------------------
// Tiny attention: hd = 2 per sub-head, 32 sub-heads, 64 batch-heads.
// One block per (bh, hh); thread = query position; K/V staged in LDS.
// ---------------------------------------------------------------------------
__global__ __launch_bounds__(256) void attn_kernel(
    const float* __restrict__ qkv, _Float16* __restrict__ attn_o)
{
  const int bh = blockIdx.x;  // 0..63
  const int hh = blockIdx.y;  // 0..31
  const int t  = threadIdx.x; // 0..255
  __shared__ float kk[GL][2];
  __shared__ float vv[GL][2];
  const float* row = qkv + ((long)bh * GL + t) * 192;
  kk[t][0] = row[64 + hh * 2];
  kk[t][1] = row[64 + hh * 2 + 1];
  vv[t][0] = row[128 + hh * 2];
  vv[t][1] = row[128 + hh * 2 + 1];
  const float q0 = row[hh * 2], q1 = row[hh * 2 + 1];
  __syncthreads();
  const float scale = 0.7071067811865476f;  // 1/sqrt(hd=2)
  float mx = -3.4e38f;
  for (int j = 0; j < GL; ++j) {
    float s = (q0 * kk[j][0] + q1 * kk[j][1]) * scale;
    mx = fmaxf(mx, s);
  }
  float sum = 0.f, o0 = 0.f, o1 = 0.f;
  for (int j = 0; j < GL; ++j) {
    float s = (q0 * kk[j][0] + q1 * kk[j][1]) * scale;
    float p = __expf(s - mx);
    sum += p;
    o0 += p * vv[j][0];
    o1 += p * vv[j][1];
  }
  float inv = 1.f / sum;
  attn_o[((long)bh * GL + t) * 64 + hh * 2]     = (_Float16)(o0 * inv);
  attn_o[((long)bh * GL + t) * 64 + hh * 2 + 1] = (_Float16)(o1 * inv);
}

// ---------------------------------------------------------------------------
// y = f + x*D ; gated by z*sigmoid(z) ; RMS-norm over 1024 ; -> f16 rows.
// ---------------------------------------------------------------------------
__global__ __launch_bounds__(256) void norm_kernel(
    const float* __restrict__ zx, const float* __restrict__ fbuf,
    const float* __restrict__ Dp, const float* __restrict__ norm_w,
    _Float16* __restrict__ yn)
{
  const int t = blockIdx.x, b = blockIdx.y, tid = threadIdx.x;
  const float* row = zx + ((long)(b * GL) + t) * GDPROJ;
  __shared__ float red[256];
  float yz[4];
  float ss = 0.f;
#pragma unroll
  for (int r = 0; r < 4; ++r) {
    int i = tid + 256 * r;
    int h = i >> 5, p = i & 31;
    float f = fbuf[((((long)b * GH) + h) * GL + t) * GP + p];
    float x = row[1024 + i];
    float y = f + x * Dp[h];
    float zv = row[i];
    float sig = 1.f / (1.f + __expf(-zv));
    float v = y * (zv * sig);
    yz[r] = v;
    ss += v * v;
  }
  red[tid] = ss;
  __syncthreads();
  for (int st = 128; st > 0; st >>= 1) {
    if (tid < st) red[tid] += red[tid + st];
    __syncthreads();
  }
  float scale = rsqrtf(red[0] / (float)GDIN + 1e-5f);
#pragma unroll
  for (int r = 0; r < 4; ++r) {
    int i = tid + 256 * r;
    yn[((long)(b * GL) + t) * GDIN + i] = (_Float16)(yz[r] * scale * norm_w[i]);
  }
}

// second output = y[:, L-1, :]
__global__ void tail_kernel(const float* __restrict__ y, float* __restrict__ o2)
{
  int i = blockIdx.x * 256 + threadIdx.x;
  if (i < GB * GDM) {
    int b = i / GDM, c = i % GDM;
    o2[i] = y[((long)(b * GL) + (GL - 1)) * GDM + c];
  }
}

// ---------------------------------------------------------------------------
// Host-side orchestration (graph-capture safe: only kernel launches).
// ---------------------------------------------------------------------------
extern "C" void kernel_launch(void* const* d_in, const int* in_sizes, int n_in,
                              void* d_out, int out_size, void* d_ws,
                              size_t ws_size, hipStream_t stream)
{
  const float* u          = (const float*)d_in[0];
  const float* in_proj_w  = (const float*)d_in[1];
  const float* dt_bias    = (const float*)d_in[2];
  const float* A_log      = (const float*)d_in[3];
  const float* Dp         = (const float*)d_in[4];
  const float* norm_w     = (const float*)d_in[5];
  const float* out_proj_w = (const float*)d_in[6];
  const float* reduce_w   = (const float*)d_in[7];
  const float* reduce_b   = (const float*)d_in[8];
  const float* attn_in_w  = (const float*)d_in[9];
  const float* attn_in_b  = (const float*)d_in[10];
  const float* attn_out_w = (const float*)d_in[11];
  const float* attn_out_b = (const float*)d_in[12];
  float* out = (float*)d_out;

  char* wsp = (char*)d_ws;
  size_t off = 0;
  auto alloc = [&](size_t bytes) -> void* {
    void* p = wsp + off;
    off = (off + bytes + 255) & ~(size_t)255;
    return p;
  };

  const long MBL = (long)GB * GL;         // 512
  const long BHL = (long)GB * GH * GL;    // 16384

  _Float16* u16    = (_Float16*)alloc(MBL * GDM * 2);
  _Float16* wi16   = (_Float16*)alloc((long)GDPROJ * GDM * 2);
  _Float16* wo16   = (_Float16*)alloc((long)GDM * GDIN * 2);
  _Float16* aiw16  = (_Float16*)alloc(192 * 64 * 2);
  _Float16* aow16  = (_Float16*)alloc(64 * 64 * 2);
  _Float16* rw16   = (_Float16*)alloc(32 * 64 * 2);
  float*    zx     = (float*)alloc(MBL * GDPROJ * 4);
  float*    Adt    = (float*)alloc(MBL * GH * 4);
  float*    xdt    = (float*)alloc(MBL * GDIN * 4);
  _Float16* Bf16   = (_Float16*)alloc(MBL * GN * 2);
  _Float16* Cf16   = (_Float16*)alloc(MBL * GN * 2);
  float*    GT     = (float*)alloc((long)GB * GL * GL * 4);
  float*    Yoff   = (float*)alloc((long)GB * GL * 2 * GDIN * 4);
  _Float16* comb16 = (_Float16*)alloc(BHL * 64 * 2);
  float*    qkv    = (float*)alloc(BHL * 192 * 4);
  _Float16* ao16   = (_Float16*)alloc(BHL * 64 * 2);
  float*    o2     = (float*)alloc(BHL * 64 * 4);
  _Float16* o216   = (_Float16*)alloc(BHL * 64 * 2);
  float*    fbuf   = (float*)alloc(BHL * 32 * 4);
  _Float16* yn16   = (_Float16*)alloc(MBL * GDIN * 2);

  auto cdiv = [](long n, long d) { return (unsigned)((n + d - 1) / d); };

  // 1) f32 -> f16 weight/activation copies
  cvt_f32_f16<<<cdiv(MBL * GDM, 256), 256, 0, stream>>>(u, u16, MBL * GDM);
  cvt_f32_f16<<<cdiv((long)GDPROJ * GDM, 256), 256, 0, stream>>>(
      in_proj_w, wi16, (long)GDPROJ * GDM);
  cvt_f32_f16<<<cdiv((long)GDM * GDIN, 256), 256, 0, stream>>>(
      out_proj_w, wo16, (long)GDM * GDIN);
  cvt_f32_f16<<<cdiv(192 * 64, 256), 256, 0, stream>>>(attn_in_w, aiw16, 192 * 64);
  cvt_f32_f16<<<cdiv(64 * 64, 256), 256, 0, stream>>>(attn_out_w, aow16, 64 * 64);
  cvt_f32_f16<<<cdiv(32 * 64, 256), 256, 0, stream>>>(reduce_w, rw16, 32 * 64);

  // 2) zxbcdt = u @ in_proj_w.T   (M=512, N=2208, K=512)
  wmma_gemm_f16<<<dim3(GDPROJ / 32, MBL / 64, 1), 256, 0, stream>>>(
      u16, wi16, nullptr, zx, MBL, GDPROJ, GDM, GDM, GDM, GDPROJ, 0, 0, 0);

  // 3) pointwise prep
  prep_kernel<<<dim3(GL, GB), 256, 0, stream>>>(zx, dt_bias, A_log, Adt, xdt,
                                                Bf16, Cf16);

  // 4) Gram matrix GT[b,j,i] = Bf[b,j].Cf[b,i]  (batched M=N=256, K=64)
  wmma_gemm_f16<<<dim3(GL / 32, GL / 64, GB), 256, 0, stream>>>(
      Bf16, Cf16, nullptr, GT, GL, GL, GN, GN, GN, GL,
      (long)GL * GN, (long)GL * GN, (long)GL * GL);

  // 5) fused cross-chunk recurrence -> Y_off windows
  yoff_kernel<<<dim3(GB, GH), 256, 0, stream>>>(GT, xdt, Adt, Yoff);

  // 6) restore + pack comb = [Yd | Yo] (f16)
  combine_kernel<<<dim3(GL, GH, GB), 64, 0, stream>>>(GT, xdt, Yoff, comb16);

  // 7) qkv = comb @ attn_in_w.T + b   (M=16384, N=192, K=64)
  wmma_gemm_f16<<<dim3(192 / 32, BHL / 64, 1), 256, 0, stream>>>(
      comb16, aiw16, attn_in_b, qkv, (int)BHL, 192, 64, 64, 64, 192, 0, 0, 0);

  // 8) attention (hd=2 per sub-head)
  attn_kernel<<<dim3(GB * GH, GH), 256, 0, stream>>>(qkv, ao16);

  // 9) o2 = attn_o @ attn_out_w.T + b   (M=16384, N=64, K=64)
  wmma_gemm_f16<<<dim3(64 / 32, BHL / 64, 1), 256, 0, stream>>>(
      ao16, aow16, attn_out_b, o2, (int)BHL, 64, 64, 64, 64, 64, 0, 0, 0);
  cvt_f32_f16<<<cdiv(BHL * 64, 256), 256, 0, stream>>>(o2, o216, BHL * 64);

  // 10) f = o2 @ reduce_w.T + b   (M=16384, N=32, K=64)
  wmma_gemm_f16<<<dim3(32 / 32, BHL / 64, 1), 256, 0, stream>>>(
      o216, rw16, reduce_b, fbuf, (int)BHL, 32, 64, 64, 64, 32, 0, 0, 0);

  // 11) residual + gate + RMS-norm -> f16 rows
  norm_kernel<<<dim3(GL, GB), 256, 0, stream>>>(zx, fbuf, Dp, norm_w, yn16);

  // 12) out = yn @ out_proj_w.T   (M=512, N=512, K=1024) -> d_out
  wmma_gemm_f16<<<dim3(GDM / 32, MBL / 64, 1), 256, 0, stream>>>(
      yn16, wo16, nullptr, out, (int)MBL, GDM, GDIN, GDIN, GDIN, GDM, 0, 0, 0);

  // 13) second output: y[:, L-1, :]
  tail_kernel<<<cdiv(GB * GDM, 256), 256, 0, stream>>>(out, out + MBL * GDM);
  (void)in_sizes; (void)n_in; (void)out_size; (void)ws_size;
}
